// WindowAttention_60378650247690
// MI455X (gfx1250) — matrix operand order repaired and checked by reading, not compiled, not run on hardware
//
#include <hip/hip_runtime.h>

typedef __attribute__((ext_vector_type(16))) _Float16 v16h;
typedef __attribute__((ext_vector_type(8)))  float    v8f;
typedef __attribute__((ext_vector_type(4)))  unsigned v4u;

#define NTOK 64
#define CDIM 128
#define HEADS 4
#define HD 32
#define NWIN 4096

// ---------------- LDS layout (bytes) ----------------
constexpr int OFF_WQ     = 0;                          // 128*128 f16 = 32768
constexpr int OFF_WKV    = OFF_WQ     + 128*128*2;     // 256*128 f16 = 65536
constexpr int OFF_WO     = OFF_WKV    + 256*128*2;     // 128*128 f16 = 32768
constexpr int OFF_BQ     = OFF_WO     + 128*128*2;     // 128 f32
constexpr int OFF_BKV    = OFF_BQ     + 128*4;         // 256 f32
constexpr int OFF_BO     = OFF_BKV    + 256*4;         // 128 f32
constexpr int OFF_BTAB   = OFF_BO     + 128*4;         // 225*4 f32 = 3600
constexpr int OFF_ATTN   = OFF_BTAB   + 900*4;         // 4*64*64 f16 = 32768 (also q/kv stage)
constexpr int OFF_QPROJ  = OFF_ATTN   + 4*64*64*2;     // 64*128 f16 = 16384 (also xbuf)
constexpr int OFF_K      = OFF_QPROJ  + 64*128*2;      // 64*128 f16 = 16384
constexpr int OFF_VT     = OFF_K      + 64*128*2;      // 128*64 f16 = 16384 (v transposed [d][tok])
constexpr int OFF_LOGITS = OFF_VT     + 128*64*2;      // 4*64*64 f32 = 65536
constexpr int OFF_MASK   = OFF_LOGITS + 4*64*64*4;     // 64*64 f32 = 16384
constexpr int SMEM_BYTES = OFF_MASK   + 64*64*4;       // 300560 B (< 320 KB WGP LDS)

#define WMMA_F16(a, b, c) \
  __builtin_amdgcn_wmma_f32_16x16x32_f16(false, (a), false, (b), (short)0, (c), false, false)

// A fragment: 16x32 f16 from row-major [m][k], row stride `stride` halfs.
// Lane l: m=l&15, g=l>>4. Halves 0-7: K=8g..8g+7 ; halves 8-15: K=16+8g..+7.
// Two 16B-aligned contiguous runs -> 2x ds_load_b128.
__device__ __forceinline__ v16h load_a_frag(const _Float16* base, int row0, int col0,
                                            int stride, int lane) {
  int m = lane & 15, g = lane >> 4;
  union { v16h h; v4u q[2]; } f;
  const _Float16* p = base + (row0 + m) * stride + col0 + 8 * g;
  f.q[0] = *(const v4u*)(p);
  f.q[1] = *(const v4u*)(p + 16);
  return f.h;
}

// B fragment: 32x16 f16 from [n][k] row-major (column-major view of B).
// Lane l: n=l&15, g=l>>4. Halves 0-7: K=8g..8g+7 ; halves 8-15: K=16+8g..+7.
__device__ __forceinline__ v16h load_b_frag(const _Float16* base, int n0, int k0,
                                            int stride, int lane) {
  int n = lane & 15, g = lane >> 4;
  union { v16h h; v4u q[2]; } f;
  const _Float16* p = base + (n0 + n) * stride + k0 + 8 * g;
  f.q[0] = *(const v4u*)(p);
  f.q[1] = *(const v4u*)(p + 16);
  return f.h;
}

// f32 global -> f16 LDS, 4 elements/thread/step (b128 load, b64 store)
__device__ __forceinline__ void stage_f16(_Float16* dst, const float* __restrict__ src,
                                          int count, int tid) {
  for (int i = tid * 4; i < count; i += 256 * 4) {
    const float4 v = *(const float4*)(src + i);
    union { _Float16 h[4]; unsigned long long u; } p;
    p.h[0] = (_Float16)v.x; p.h[1] = (_Float16)v.y;
    p.h[2] = (_Float16)v.z; p.h[3] = (_Float16)v.w;
    *(unsigned long long*)(dst + i) = p.u;
  }
}

__global__ __launch_bounds__(256, 1)
void winattn_fused_kernel(const float* __restrict__ q, const float* __restrict__ kv,
                          const float* __restrict__ mask,
                          const float* __restrict__ Wq, const float* __restrict__ bq,
                          const float* __restrict__ Wkv, const float* __restrict__ bkv,
                          const float* __restrict__ Wo, const float* __restrict__ bo,
                          const float* __restrict__ btab,
                          float* __restrict__ out) {
  extern __shared__ char smem[];
  _Float16* wq_s    = (_Float16*)(smem + OFF_WQ);
  _Float16* wkv_s   = (_Float16*)(smem + OFF_WKV);
  _Float16* wo_s    = (_Float16*)(smem + OFF_WO);
  float*    bq_s    = (float*)(smem + OFF_BQ);
  float*    bkv_s   = (float*)(smem + OFF_BKV);
  float*    bo_s    = (float*)(smem + OFF_BO);
  float*    btab_s  = (float*)(smem + OFF_BTAB);
  _Float16* attn_s  = (_Float16*)(smem + OFF_ATTN);   // phase-D output
  _Float16* stage_q = attn_s;                          // dead after phase B
  _Float16* stage_kv= attn_s + 64 * 128;
  _Float16* qproj_s = (_Float16*)(smem + OFF_QPROJ);   // xbuf in phase E
  _Float16* k_s     = (_Float16*)(smem + OFF_K);
  _Float16* vt_s    = (_Float16*)(smem + OFF_VT);
  float*    logit_s = (float*)(smem + OFF_LOGITS);
  float*    mask_s  = (float*)(smem + OFF_MASK);

  const int b    = blockIdx.x;
  const int w    = b & (NWIN - 1);
  const int tid  = threadIdx.x;
  const int wave = __builtin_amdgcn_readfirstlane(tid >> 5);  // scalar wave id
  const int lane = tid & 31;
  const int nn   = lane & 15;
  const int g    = lane >> 4;

  // ---------------- Phase A: cooperative staging ----------------
  stage_f16(wq_s,  Wq,  128 * 128, tid);
  stage_f16(wkv_s, Wkv, 256 * 128, tid);
  stage_f16(wo_s,  Wo,  128 * 128, tid);
  if (tid < 128) { bq_s[tid] = bq[tid]; bo_s[tid] = bo[tid]; }
  if (tid < 256) bkv_s[tid] = bkv[tid];
  if (tid * 4 < 900)
    *(float4*)(btab_s + tid * 4) = *(const float4*)(btab + tid * 4);
  stage_f16(stage_q,  q  + (size_t)b * (NTOK * CDIM), NTOK * CDIM, tid);
  stage_f16(stage_kv, kv + (size_t)b * (NTOK * CDIM), NTOK * CDIM, tid);
  {
    const float* mg = mask + (size_t)w * (NTOK * NTOK);
    for (int i = tid * 4; i < NTOK * NTOK; i += 256 * 4)
      *(float4*)(mask_s + i) = *(const float4*)(mg + i);
  }
  __syncthreads();

  // ---------------- Phase B: Q and KV projections ----------------
  {  // Q: wave owns N-tile ni = wave; hoist 4 weight B-fragments, loop M.
    const int ni = wave;
    v16h bw[4];
#pragma unroll
    for (int kk = 0; kk < 4; ++kk)
      bw[kk] = load_b_frag(wq_s, ni * 16, kk * 32, CDIM, lane);
    const float bb = bq_s[ni * 16 + nn];
#pragma unroll
    for (int mi = 0; mi < 4; ++mi) {
      v8f c = {bb, bb, bb, bb, bb, bb, bb, bb};
#pragma unroll
      for (int kk = 0; kk < 4; ++kk) {
        v16h a = load_a_frag(stage_q, mi * 16, kk * 32, CDIM, lane);
        c = WMMA_F16(a, bw[kk], c);
      }
#pragma unroll
      for (int r = 0; r < 8; ++r)
        qproj_s[(mi * 16 + r + 8 * g) * CDIM + ni * 16 + nn] = (_Float16)c[r];
    }
  }
  // KV: wave owns N-tiles ni = wave (K half) and wave+8 (V half).
#pragma unroll
  for (int half = 0; half < 2; ++half) {
    const int ni = wave + half * 8;
    v16h bw[4];
#pragma unroll
    for (int kk = 0; kk < 4; ++kk)
      bw[kk] = load_b_frag(wkv_s, ni * 16, kk * 32, CDIM, lane);
    const float bb = bkv_s[ni * 16 + nn];
#pragma unroll
    for (int mi = 0; mi < 4; ++mi) {
      v8f c = {bb, bb, bb, bb, bb, bb, bb, bb};
#pragma unroll
      for (int kk = 0; kk < 4; ++kk) {
        v16h a = load_a_frag(stage_kv, mi * 16, kk * 32, CDIM, lane);
        c = WMMA_F16(a, bw[kk], c);
      }
      if (half == 0) {   // K half: natural [tok][ch]
#pragma unroll
        for (int r = 0; r < 8; ++r)
          k_s[(mi * 16 + r + 8 * g) * CDIM + ni * 16 + nn] = (_Float16)c[r];
      } else {           // V half: transposed [ch][tok] (contiguous over r -> b128)
        const int ch = (ni - 8) * 16 + nn;
#pragma unroll
        for (int r = 0; r < 8; ++r)
          vt_s[ch * NTOK + mi * 16 + r + 8 * g] = (_Float16)c[r];
      }
    }
  }
  __syncthreads();

  // -------- Phase C: logits = (q . k^T) * scale + mask (mask folded here) --------
  const float scale = 0.17677669529663687f;  // 1/sqrt(32)
  {
    const int h   = wave >> 1;
    const int ni0 = (wave & 1) * 2;     // this wave does key tiles ni0, ni0+1
    v16h b0 = load_b_frag(k_s, ni0 * 16,       h * HD, CDIM, lane);
    v16h b1 = load_b_frag(k_s, (ni0 + 1) * 16, h * HD, CDIM, lane);
    float* lbase = logit_s + h * (NTOK * NTOK);
#pragma unroll
    for (int mi = 0; mi < 4; ++mi) {
      v16h a = load_a_frag(qproj_s, mi * 16, h * HD, CDIM, lane);
      v8f c0 = {}; c0 = WMMA_F16(a, b0, c0);
      v8f c1 = {}; c1 = WMMA_F16(a, b1, c1);
#pragma unroll
      for (int r = 0; r < 8; ++r) {
        const int o = (mi * 16 + r + 8 * g) * NTOK + ni0 * 16 + nn;
        lbase[o]      = c0[r] * scale + mask_s[o];
        lbase[o + 16] = c1[r] * scale + mask_s[o + 16];
      }
    }
  }
  __syncthreads();

  // ---------- Phase D: +rel_bias, softmax, -> f16 (row held in registers) ----------
  {
    const int h = tid >> 6, row = tid & 63;   // 4 heads * 64 rows = 256 threads
    const float* lrow = logit_s + h * (NTOK * NTOK) + row * NTOK;
    const int i1 = row >> 3, j1 = row & 7;
    float4 vals[16];
    float mx = -1e30f;
#pragma unroll
    for (int c = 0; c < 16; ++c) {
      float4 v = *(const float4*)(lrow + c * 4);
      const int m0 = c * 4;
      const int i2 = m0 >> 3;
      // rel-pos index is affine in m within a 4-chunk: idx, idx-1, idx-2, idx-3
      const int base = (i1 - i2 + 7) * 15 + (j1 - (m0 & 7) + 7);
      v.x += btab_s[(base    ) * 4 + h];
      v.y += btab_s[(base - 1) * 4 + h];
      v.z += btab_s[(base - 2) * 4 + h];
      v.w += btab_s[(base - 3) * 4 + h];
      vals[c] = v;
      mx = fmaxf(mx, fmaxf(fmaxf(v.x, v.y), fmaxf(v.z, v.w)));
    }
    float s = 0.f;
#pragma unroll
    for (int c = 0; c < 16; ++c) {
      float4 v = vals[c];
      v.x = __expf(v.x - mx); v.y = __expf(v.y - mx);
      v.z = __expf(v.z - mx); v.w = __expf(v.w - mx);
      vals[c] = v;
      s += (v.x + v.y) + (v.z + v.w);
    }
    const float inv = 1.0f / s;
    _Float16* arow = attn_s + h * (NTOK * NTOK) + row * NTOK;
#pragma unroll
    for (int c = 0; c < 16; ++c) {
      union { _Float16 hh[4]; unsigned long long u; } p;
      p.hh[0] = (_Float16)(vals[c].x * inv);
      p.hh[1] = (_Float16)(vals[c].y * inv);
      p.hh[2] = (_Float16)(vals[c].z * inv);
      p.hh[3] = (_Float16)(vals[c].w * inv);
      *(unsigned long long*)(arow + c * 4) = p.u;
    }
  }
  __syncthreads();

  // ---------------- Phase E: x = attn @ v  (per head) ----------------
  _Float16* xbuf = qproj_s;  // reuse
  {
    const int h = wave >> 1, nj = wave & 1;
    v16h b0 = load_b_frag(vt_s, h * HD + nj * 16, 0,  NTOK, lane);
    v16h b1 = load_b_frag(vt_s, h * HD + nj * 16, 32, NTOK, lane);
    const _Float16* abase = attn_s + h * (NTOK * NTOK);
#pragma unroll
    for (int mi = 0; mi < 4; ++mi) {
      v16h a0 = load_a_frag(abase, mi * 16, 0,  NTOK, lane);
      v16h a1 = load_a_frag(abase, mi * 16, 32, NTOK, lane);
      v8f c = {};
      c = WMMA_F16(a0, b0, c);
      c = WMMA_F16(a1, b1, c);
#pragma unroll
      for (int r = 0; r < 8; ++r)
        xbuf[(mi * 16 + r + 8 * g) * CDIM + h * HD + nj * 16 + nn] = (_Float16)c[r];
    }
  }
  __syncthreads();

  // ---------------- Phase F: out = x @ Wo^T + bo ----------------
  float* og = out + (size_t)b * (NTOK * CDIM);
  {
    const int ni = wave;
    v16h bw[4];
#pragma unroll
    for (int kk = 0; kk < 4; ++kk)
      bw[kk] = load_b_frag(wo_s, ni * 16, kk * 32, CDIM, lane);
    const float bb = bo_s[ni * 16 + nn];
#pragma unroll
    for (int mi = 0; mi < 4; ++mi) {
      v8f c = {bb, bb, bb, bb, bb, bb, bb, bb};
#pragma unroll
      for (int kk = 0; kk < 4; ++kk) {
        v16h a = load_a_frag(xbuf, mi * 16, kk * 32, CDIM, lane);
        c = WMMA_F16(a, bw[kk], c);
      }
#pragma unroll
      for (int r = 0; r < 8; ++r)
        og[(mi * 16 + r + 8 * g) * CDIM + ni * 16 + nn] = c[r];
    }
  }
}

extern "C" void kernel_launch(void* const* d_in, const int* in_sizes, int n_in,
                              void* d_out, int out_size, void* d_ws, size_t ws_size,
                              hipStream_t stream) {
  const float* q    = (const float*)d_in[0];
  const float* kv   = (const float*)d_in[1];
  const float* mask = (const float*)d_in[2];
  const float* Wq   = (const float*)d_in[3];
  const float* bq   = (const float*)d_in[4];
  const float* Wkv  = (const float*)d_in[5];
  const float* bkv  = (const float*)d_in[6];
  const float* Wo   = (const float*)d_in[7];
  const float* bo   = (const float*)d_in[8];
  const float* btab = (const float*)d_in[9];
  float* out = (float*)d_out;

  const int B_ = 8192;
  winattn_fused_kernel<<<dim3(B_), dim3(256), SMEM_BYTES, stream>>>(
      q, kv, mask, Wq, bq, Wkv, bkv, Wo, bo, btab, out);
}
